// GMNPropagator_8924942041966
// MI455X (gfx1250) — matrix operand order repaired and checked by reading, not compiled, not run on hardware
//
#include <hip/hip_runtime.h>

// ---------------------------------------------------------------------------
// GMN propagator for MI455X (gfx1250, wave32, WMMA bf16 -> f32).
//   0) memsetAsync m_sum = 0
//   1) prep_nodes : x->bf16, row-normalize (n1,n2 bf16), x1T/x2T (bf16, feat-major)
//   2) prep_wT x4 : weights -> bf16, transposed (output-channel-major)
//   3) msg_kernel : per-WG 16 edges, gather->LDS, 256->256 relu -> LDS,
//                   256->256, f32 global atomicAdd scatter into m_sum
//   4) attn_kernel x2 : transposed flash attention; K/V tiles staged in LDS
//                   once per workgroup via TDM tensor_load_to_lds (double
//                   buffered, s_wait_tensorcnt) with a cooperative-copy
//                   fallback; online softmax over 12288 keys; u = x - (P@V)/l
//   5) node_kernel: per-WG 16 nodes, concat(x,m_sum,u)->LDS bf16,
//                   512->512 relu -> LDS, 512->128 -> d_out (f32)
// ---------------------------------------------------------------------------

#define M_NODES 12288
#define N_NODES 12288
#define TOT     24576
#define D       128
#define OUTC    128
#define EDG     196608
#define EDG2    (2*EDG)

typedef __attribute__((ext_vector_type(16))) __bf16 v16bf;
typedef __attribute__((ext_vector_type(8)))  float  v8f;

#if __has_builtin(__builtin_amdgcn_tensor_load_to_lds) && \
    __has_builtin(__builtin_amdgcn_s_wait_tensorcnt)
#define USE_TDM 1
#if __clang_major__ >= 23
#define TDM_6ARG 1
#endif
typedef unsigned int v4ui __attribute__((ext_vector_type(4)));
typedef int          v4i  __attribute__((ext_vector_type(4)));
typedef int          v8i  __attribute__((ext_vector_type(8)));
#endif

__device__ __forceinline__ v8f wmma_bf16(v16bf a, v16bf b, v8f c) {
  // D = A(16x32 bf16) x B(32x16 bf16) + C(16x16 f32)
  return __builtin_amdgcn_wmma_f32_16x16x32_bf16(false, a, false, b, (short)0, c,
                                                 false, false);
}

// A operand, 16x32 bf16. Caller passes &src[m*ld + k0 + bh*16] where m=lane&15,
// bh=lane>>4. Element e <-> k = bh*16 + e (contiguous, matches ldB16).
__device__ __forceinline__ v16bf ldA16(const __bf16* p) {
  v16bf a;
#pragma unroll
  for (int i = 0; i < 8; ++i) { a[i] = p[i]; a[i + 8] = p[8 + i]; }
  return a;
}

// B operand, 32x16 bf16. Caller passes &src[col_row_base + k0 + bh*16]:
// lane half bh covers K = bh*16 .. bh*16+15, element e <-> k = bh*16 + e.
__device__ __forceinline__ v16bf ldB16(const __bf16* p) {
  v16bf b;
#pragma unroll
  for (int i = 0; i < 16; ++i) b[i] = p[i];
  return b;
}

__device__ __forceinline__ v8f vzero8() {
  v8f z = {0.f, 0.f, 0.f, 0.f, 0.f, 0.f, 0.f, 0.f};
  return z;
}

#ifdef USE_TDM
// Issue a 2D TDM tile load: height rows x width elems (bf16), row pitch
// stride elems, from gaddr into LDS byte offset lds_addr. D# packing per
// CDNA5 ISA 8.3/8.4 (group0: count/lds/global/type; group1: data_size,
// tensor_dim0/1, tile_dim0/1, tensor_dim0_stride). Wave-level op, EXEC
// ignored; tracked by TENSORcnt.
__device__ __forceinline__ void tdm_load_2d(unsigned lds_addr, const void* gaddr,
                                            unsigned width, unsigned height,
                                            unsigned stride) {
  unsigned long long ga = (unsigned long long)gaddr;
  v4ui g0;
  g0[0] = 1u;                                          // count=1 (user, load)
  g0[1] = lds_addr;                                    // lds_addr[31:0]
  g0[2] = (unsigned)(ga & 0xFFFFFFFFu);                // global_addr[31:0]
  g0[3] = (unsigned)((ga >> 32) & 0x01FFFFFFu)         // global_addr[56:32]
          | (2u << 30);                                // type=2 ("image")
  v8i g1;
  g1[0] = (int)(1u << 16);                             // data_size=1 (2 bytes)
  g1[1] = (int)((width & 0xFFFFu) << 16);              // tensor_dim0[15:0]
  g1[2] = (int)((width >> 16) | ((height & 0xFFFFu) << 16)); // td0 hi | td1 lo
  g1[3] = (int)((height >> 16) | (width << 16));       // td1 hi | tile_dim0
  g1[4] = (int)(height & 0xFFFFu);                     // tile_dim1 (tile_dim2=0)
  g1[5] = (int)stride;                                 // tensor_dim0_stride lo32
  g1[6] = 0;                                           // stride hi | td1_stride lo
  g1[7] = 0;
  v4i z = {0, 0, 0, 0};
#ifdef TDM_6ARG
  v8i z8 = {0, 0, 0, 0, 0, 0, 0, 0};
  __builtin_amdgcn_tensor_load_to_lds(g0, g1, z, z, z8, 0);
#else
  __builtin_amdgcn_tensor_load_to_lds(g0, g1, z, z, 0);
#endif
}
#endif

// ---------------------------------------------------------------------------
// prep_nodes: one wave per node row (8 rows / 256-thread block).
// ---------------------------------------------------------------------------
__global__ void __launch_bounds__(256)
prep_nodes(const float* __restrict__ x1, const float* __restrict__ x2,
           __bf16* __restrict__ xb,
           __bf16* __restrict__ n1b, __bf16* __restrict__ n2b,
           __bf16* __restrict__ x1T, __bf16* __restrict__ x2T) {
  int lane = threadIdx.x & 31;
  int row  = blockIdx.x * 8 + (threadIdx.x >> 5);
  bool g1  = row < M_NODES;
  int  rl  = g1 ? row : row - M_NODES;
  const float* src = g1 ? (x1 + (size_t)rl * D) : (x2 + (size_t)rl * D);
  float v[4];
  float ss = 0.f;
#pragma unroll
  for (int i = 0; i < 4; ++i) { v[i] = src[lane * 4 + i]; ss += v[i] * v[i]; }
#pragma unroll
  for (int o = 16; o >= 1; o >>= 1) ss += __shfl_xor(ss, o, 32);
  float inv = 1.0f / fmaxf(sqrtf(ss), 1e-12f);
  __bf16* nd = g1 ? n1b : n2b;
  __bf16* td = g1 ? x1T : x2T;
  int cnt = g1 ? M_NODES : N_NODES;
#pragma unroll
  for (int i = 0; i < 4; ++i) {
    int c = lane * 4 + i;
    xb[(size_t)row * D + c] = (__bf16)v[i];
    nd[(size_t)rl  * D + c] = (__bf16)(v[i] * inv);
    td[(size_t)c * cnt + rl] = (__bf16)v[i];   // V^T[f][node]
  }
}

// WT[c*K + k] = (bf16) W[k*C + c]
__global__ void __launch_bounds__(256)
prep_wT(const float* __restrict__ W, __bf16* __restrict__ WT, int K, int C) {
  int idx = blockIdx.x * 256 + threadIdx.x;
  if (idx >= K * C) return;
  int c = idx / K, k = idx - c * K;
  WT[(size_t)c * K + k] = (__bf16)W[(size_t)k * C + c];
}

// ---------------------------------------------------------------------------
// msg_kernel: 16 edges / 256-thread workgroup (8 waves). Fused 2-layer MLP,
// f32 global atomic scatter into m_sum.
// ---------------------------------------------------------------------------
__global__ void __launch_bounds__(256)
msg_kernel(const __bf16* __restrict__ xb,
           const int* __restrict__ e1, const int* __restrict__ e2,
           const __bf16* __restrict__ W1T, const float* __restrict__ b1,
           const __bf16* __restrict__ W2T, const float* __restrict__ b2,
           float* __restrict__ msum) {
  __shared__ __bf16 Af[16 * 256];
  __shared__ __bf16 Hs[16 * 256];
  __shared__ int Rn[16];
  int t  = threadIdx.x;
  int g0 = blockIdx.x * 16;
  {
    int e = t >> 4, p = t & 15;
    int g = g0 + e;
    int r, c;
    if (g < EDG) { r = e1[g]; c = e1[EDG + g]; }
    else { int gg = g - EDG; r = e2[gg] + M_NODES; c = e2[EDG + gg] + M_NODES; }
    if (p == 0) Rn[e] = r;
    int node = (p < 8) ? r : c;
    const uint4* s = (const uint4*)&xb[(size_t)node * D + (p & 7) * 16];
    uint4* dst = (uint4*)&Af[e * 256 + p * 16];
    dst[0] = s[0]; dst[1] = s[1];
  }
  __syncthreads();

  int lane = t & 31, w = t >> 5;
  int n = lane & 15, bh = lane >> 4;
  int c0 = w * 32;

  v8f a0 = vzero8(), a1 = vzero8();
#pragma unroll
  for (int ks = 0; ks < 8; ++ks) {
    int k0 = ks * 32;
    v16bf A  = ldA16(&Af[n * 256 + k0 + bh * 16]);
    v16bf B0 = ldB16(&W1T[(size_t)(c0 + n) * 256 + k0 + bh * 16]);
    a0 = wmma_bf16(A, B0, a0);
    v16bf B1 = ldB16(&W1T[(size_t)(c0 + 16 + n) * 256 + k0 + bh * 16]);
    a1 = wmma_bf16(A, B1, a1);
  }
#pragma unroll
  for (int v = 0; v < 8; ++v) {
    int row = v + 8 * bh;
    Hs[row * 256 + c0 + n]      = (__bf16)fmaxf(a0[v] + b1[c0 + n], 0.f);
    Hs[row * 256 + c0 + 16 + n] = (__bf16)fmaxf(a1[v] + b1[c0 + 16 + n], 0.f);
  }
  __syncthreads();

  a0 = vzero8(); a1 = vzero8();
#pragma unroll
  for (int ks = 0; ks < 8; ++ks) {
    int k0 = ks * 32;
    v16bf A  = ldA16(&Hs[n * 256 + k0 + bh * 16]);
    v16bf B0 = ldB16(&W2T[(size_t)(c0 + n) * 256 + k0 + bh * 16]);
    a0 = wmma_bf16(A, B0, a0);
    v16bf B1 = ldB16(&W2T[(size_t)(c0 + 16 + n) * 256 + k0 + bh * 16]);
    a1 = wmma_bf16(A, B1, a1);
  }
#pragma unroll
  for (int v = 0; v < 8; ++v) {
    int row = v + 8 * bh;
    int rn  = Rn[row];
    atomicAdd(&msum[(size_t)rn * 256 + c0 + n],      a0[v] + b2[c0 + n]);
    atomicAdd(&msum[(size_t)rn * 256 + c0 + 16 + n], a1[v] + b2[c0 + 16 + n]);
  }
}

// ---------------------------------------------------------------------------
// attn_kernel: transposed flash attention; one 16-query tile per wave; the
// whole workgroup shares a double-buffered LDS stage of each 32-key K-tile
// (32x128 bf16) and V^T-slice (128x32 bf16), loaded by the Tensor Data Mover
// (wave 0 issues tensor_load_to_lds, waits TENSORcnt, barrier publishes) or
// by a cooperative copy when the TDM builtin is unavailable.
// ---------------------------------------------------------------------------
__global__ void __launch_bounds__(256)
attn_kernel(const __bf16* __restrict__ nQ, const __bf16* __restrict__ nK,
            const __bf16* __restrict__ VT, const float* __restrict__ xq,
            __bf16* __restrict__ uout, int NK) {
  __shared__ __bf16 Kb[2][32 * 128];   // keys tile, row-major [key][feat]
  __shared__ __bf16 Vb[2][128 * 32];   // V^T tile, row-major [feat][key]
  int t = threadIdx.x;
  int lane = t & 31, wid = t >> 5;
  int n = lane & 15, bh = lane >> 4;
  int q0 = (blockIdx.x * 8 + wid) * 16;

  // Hoisted query B-operands: 4 K-chunks of nQ tile (loop-invariant).
  v16bf Bq[4];
#pragma unroll
  for (int kc = 0; kc < 4; ++kc)
    Bq[kc] = ldB16(&nQ[(size_t)(q0 + n) * D + kc * 32 + bh * 16]);

  v8f O[8];
#pragma unroll
  for (int tt = 0; tt < 8; ++tt) O[tt] = vzero8();
  float mrun = -3.0e38f, lrun = 0.f;

#ifdef USE_TDM
  if (wid == 0) {
    tdm_load_2d((unsigned)(size_t)&Kb[0][0], &nK[0], 128u, 32u, 128u);
    tdm_load_2d((unsigned)(size_t)&Vb[0][0], &VT[0], 32u, 128u, (unsigned)NK);
    __builtin_amdgcn_s_wait_tensorcnt(0);
  }
#else
  {
    // cooperative stage of tile j0=0 into buffer 0
    int r = t >> 3, c8 = (t & 7) * 16;
    const uint4* ks = (const uint4*)&nK[(size_t)r * D + c8];
    uint4* kd = (uint4*)&Kb[0][r * 128 + c8];
    kd[0] = ks[0]; kd[1] = ks[1];
    int f = t >> 1, p = (t & 1) * 16;
    const uint4* vs = (const uint4*)&VT[(size_t)f * NK + p];
    uint4* vd = (uint4*)&Vb[0][f * 32 + p];
    vd[0] = vs[0]; vd[1] = vs[1];
  }
#endif
  __syncthreads();

  for (int j0 = 0; j0 < NK; j0 += 32) {
    int cur = (j0 >> 5) & 1, nxt = cur ^ 1;
    if (j0 + 32 < NK) {
      int jn = j0 + 32;
#ifdef USE_TDM
      if (wid == 0) {
        tdm_load_2d((unsigned)(size_t)&Kb[nxt][0], &nK[(size_t)jn * D],
                    128u, 32u, 128u);
        tdm_load_2d((unsigned)(size_t)&Vb[nxt][0], &VT[jn],
                    32u, 128u, (unsigned)NK);
      }
#else
      int r = t >> 3, c8 = (t & 7) * 16;
      const uint4* ks = (const uint4*)&nK[(size_t)(jn + r) * D + c8];
      uint4* kd = (uint4*)&Kb[nxt][r * 128 + c8];
      kd[0] = ks[0]; kd[1] = ks[1];
      int f = t >> 1, p = (t & 1) * 16;
      const uint4* vs = (const uint4*)&VT[(size_t)f * NK + jn + p];
      uint4* vd = (uint4*)&Vb[nxt][f * 32 + p];
      vd[0] = vs[0]; vd[1] = vs[1];
#endif
      if (j0 + 64 < NK)
        __builtin_prefetch(&nK[(size_t)(j0 + 64) * D], 0, 1);
    }

    // logits: two 16-key tiles, K = 128 in 4 WMMA steps each (A from LDS)
    v8f S0 = vzero8(), S1 = vzero8();
#pragma unroll
    for (int kc = 0; kc < 4; ++kc) {
      v16bf A0 = ldA16(&Kb[cur][n * 128 + kc * 32 + bh * 16]);
      S0 = wmma_bf16(A0, Bq[kc], S0);
      v16bf A1 = ldA16(&Kb[cur][(16 + n) * 128 + kc * 32 + bh * 16]);
      S1 = wmma_bf16(A1, Bq[kc], S1);
    }

    // online softmax along keys (VGPR axis + cross-half xor)
    float mx = -3.0e38f;
#pragma unroll
    for (int v = 0; v < 8; ++v) { mx = fmaxf(mx, S0[v]); mx = fmaxf(mx, S1[v]); }
    mx = fmaxf(mx, __shfl_xor(mx, 16, 32));
    float mnew  = fmaxf(mrun, mx);
    float scale = __expf(mrun - mnew);
    float p0[8], p1[8], rs = 0.f;
#pragma unroll
    for (int v = 0; v < 8; ++v) {
      p0[v] = __expf(S0[v] - mnew);
      p1[v] = __expf(S1[v] - mnew);
      rs += p0[v] + p1[v];
    }
    rs += __shfl_xor(rs, 16, 32);
    lrun = lrun * scale + rs;
    mrun = mnew;
#pragma unroll
    for (int tt = 0; tt < 8; ++tt)
#pragma unroll
      for (int v = 0; v < 8; ++v) O[tt][v] *= scale;

    // C-layout (S^T) -> B-operand layout for P^T (32 keys x 16 queries):
    // element e (k = bh*16+e); keys 8..15 of a tile live in the opposite
    // lane half -> shfl_xor(16).
    v16bf BP;
#pragma unroll
    for (int v = 0; v < 8; ++v) {
      float s0 = __shfl_xor(p0[v], 16, 32);
      float s1 = __shfl_xor(p1[v], 16, 32);
      float lo = bh ? s1 : p0[v];
      float hi = bh ? p1[v] : s0;
      BP[v]     = (__bf16)lo;
      BP[8 + v] = (__bf16)hi;
    }

    // O^T += V^T(16 feats x 32 keys) @ P^T  for each of 8 feature tiles
#pragma unroll
    for (int tt = 0; tt < 8; ++tt) {
      v16bf Av = ldA16(&Vb[cur][(tt * 16 + n) * 32 + bh * 16]);
      O[tt] = wmma_bf16(Av, BP, O[tt]);
    }

#ifdef USE_TDM
    if (wid == 0) __builtin_amdgcn_s_wait_tensorcnt(0);
#endif
    __syncthreads();
  }

  float inv = 1.f / lrun;
  int q = q0 + n;
#pragma unroll
  for (int tt = 0; tt < 8; ++tt)
#pragma unroll
    for (int v = 0; v < 8; ++v) {
      int f = tt * 16 + v + 8 * bh;
      float u = xq[(size_t)q * D + f] - O[tt][v] * inv;
      uout[(size_t)q * D + f] = (__bf16)u;
    }
}

// ---------------------------------------------------------------------------
// node_kernel: 16 nodes / workgroup. concat(x, m_sum, u) -> 512 bf16 in LDS,
// fused 512->512 relu -> 512->128, f32 result straight to d_out.
// ---------------------------------------------------------------------------
__global__ void __launch_bounds__(256)
node_kernel(const __bf16* __restrict__ xb, const float* __restrict__ msum,
            const __bf16* __restrict__ ub,
            const __bf16* __restrict__ W1T, const float* __restrict__ b1,
            const __bf16* __restrict__ W2T, const float* __restrict__ b2,
            float* __restrict__ out) {
  __shared__ __bf16 Xc[16 * 512];
  __shared__ __bf16 Hs[16 * 512];
  int t   = threadIdx.x;
  int nd0 = blockIdx.x * 16;
  {
    int e = t >> 4, p = t & 15;  // each thread fills 32 bf16 of the 512 concat
    int node = nd0 + e;
    if (p < 4) {
      const uint4* s = (const uint4*)&xb[(size_t)node * D + p * 32];
      uint4* dd = (uint4*)&Xc[e * 512 + p * 32];
      dd[0] = s[0]; dd[1] = s[1]; dd[2] = s[2]; dd[3] = s[3];
    } else if (p < 12) {
      const float* s = &msum[(size_t)node * 256 + (p - 4) * 32];
#pragma unroll
      for (int i = 0; i < 32; ++i) Xc[e * 512 + p * 32 + i] = (__bf16)s[i];
    } else {
      const uint4* s = (const uint4*)&ub[(size_t)node * D + (p - 12) * 32];
      uint4* dd = (uint4*)&Xc[e * 512 + p * 32];
      dd[0] = s[0]; dd[1] = s[1]; dd[2] = s[2]; dd[3] = s[3];
    }
  }
  __syncthreads();

  int lane = t & 31, w = t >> 5;
  int n = lane & 15, bh = lane >> 4;
  {
    int c0 = w * 64;
    v8f acc[4];
#pragma unroll
    for (int nt = 0; nt < 4; ++nt) acc[nt] = vzero8();
#pragma unroll
    for (int ks = 0; ks < 16; ++ks) {
      int k0 = ks * 32;
      v16bf A = ldA16(&Xc[n * 512 + k0 + bh * 16]);
#pragma unroll
      for (int nt = 0; nt < 4; ++nt) {
        v16bf B = ldB16(&W1T[(size_t)(c0 + nt * 16 + n) * 512 + k0 + bh * 16]);
        acc[nt] = wmma_bf16(A, B, acc[nt]);
      }
    }
#pragma unroll
    for (int nt = 0; nt < 4; ++nt)
#pragma unroll
      for (int v = 0; v < 8; ++v) {
        int row = v + 8 * bh, col = c0 + nt * 16 + n;
        Hs[row * 512 + col] = (__bf16)fmaxf(acc[nt][v] + b1[col], 0.f);
      }
  }
  __syncthreads();
  {
    int c0 = w * 16;
    v8f acc = vzero8();
#pragma unroll
    for (int ks = 0; ks < 16; ++ks) {
      int k0 = ks * 32;
      v16bf A = ldA16(&Hs[n * 512 + k0 + bh * 16]);
      v16bf B = ldB16(&W2T[(size_t)(c0 + n) * 512 + k0 + bh * 16]);
      acc = wmma_bf16(A, B, acc);
    }
#pragma unroll
    for (int v = 0; v < 8; ++v) {
      int row = v + 8 * bh;
      out[(size_t)(nd0 + row) * OUTC + c0 + n] = acc[v] + b2[c0 + n];
    }
  }
}

// ---------------------------------------------------------------------------
extern "C" void kernel_launch(void* const* d_in, const int* in_sizes, int n_in,
                              void* d_out, int out_size, void* d_ws, size_t ws_size,
                              hipStream_t stream) {
  const float* x1  = (const float*)d_in[0];
  const int*   e1  = (const int*)  d_in[1];
  const float* x2  = (const float*)d_in[2];
  const int*   e2  = (const int*)  d_in[3];
  const float* Wm1 = (const float*)d_in[4];
  const float* bm1 = (const float*)d_in[5];
  const float* Wm2 = (const float*)d_in[6];
  const float* bm2 = (const float*)d_in[7];
  const float* Wn1 = (const float*)d_in[8];
  const float* bn1 = (const float*)d_in[9];
  const float* Wn2 = (const float*)d_in[10];
  const float* bn2 = (const float*)d_in[11];
  float* out = (float*)d_out;

  char* ws = (char*)d_ws;
  size_t off = 0;
  auto take = [&](size_t bytes) -> char* {
    char* p = ws + off;
    off = (off + bytes + 255) & ~(size_t)255;
    return p;
  };
  __bf16* xb   = (__bf16*)take((size_t)TOT * D * 2);
  __bf16* n1b  = (__bf16*)take((size_t)M_NODES * D * 2);
  __bf16* n2b  = (__bf16*)take((size_t)N_NODES * D * 2);
  __bf16* x1T  = (__bf16*)take((size_t)D * M_NODES * 2);
  __bf16* x2T  = (__bf16*)take((size_t)D * N_NODES * 2);
  __bf16* ubuf = (__bf16*)take((size_t)TOT * D * 2);
  float*  msum = (float*) take((size_t)TOT * 256 * 4);
  __bf16* w1T  = (__bf16*)take((size_t)256 * 256 * 2);
  __bf16* w2T  = (__bf16*)take((size_t)256 * 256 * 2);
  __bf16* wn1T = (__bf16*)take((size_t)512 * 512 * 2);
  __bf16* wn2T = (__bf16*)take((size_t)128 * 512 * 2);

  hipMemsetAsync(msum, 0, (size_t)TOT * 256 * 4, stream);

  prep_nodes<<<TOT / 8, 256, 0, stream>>>(x1, x2, xb, n1b, n2b, x1T, x2T);
  prep_wT<<<(256 * 256 + 255) / 256, 256, 0, stream>>>(Wm1, w1T, 256, 256);
  prep_wT<<<(256 * 256 + 255) / 256, 256, 0, stream>>>(Wm2, w2T, 256, 256);
  prep_wT<<<(512 * 512 + 255) / 256, 256, 0, stream>>>(Wn1, wn1T, 512, 512);
  prep_wT<<<(512 * 128 + 255) / 256, 256, 0, stream>>>(Wn2, wn2T, 512, 128);

  msg_kernel<<<EDG2 / 16, 256, 0, stream>>>(xb, e1, e2, w1T, bm1, w2T, bm2, msum);

  attn_kernel<<<M_NODES / 128, 256, 0, stream>>>(n1b, n2b, x2T, x1, ubuf, N_NODES);
  attn_kernel<<<N_NODES / 128, 256, 0, stream>>>(n2b, n1b, x1T, x2,
                                                 ubuf + (size_t)M_NODES * D, M_NODES);

  node_kernel<<<TOT / 16, 256, 0, stream>>>(xb, msum, ubuf, wn1T, bn1, wn2T, bn2, out);
}
